// DCNv2PackFlowGuided_12498354831829
// MI455X (gfx1250) — matrix operand order repaired and checked by reading, not compiled, not run on hardware
//
#include <hip/hip_runtime.h>
#include <hip/hip_bf16.h>

typedef __attribute__((ext_vector_type(16))) _Float16 v16h;
typedef __attribute__((ext_vector_type(8)))  _Float16 v8h;
typedef __attribute__((ext_vector_type(8)))  float    v8f;

static constexpr int B_   = 4;
static constexpr int C_   = 64;
static constexpr int H_   = 128;
static constexpr int W_   = 128;
static constexpr int HWp  = H_ * W_;          // pixels per channel plane
static constexpr int CIN1 = 130;              // 64 + 64 + 2
static constexpr int K1   = CIN1 * 9;         // 1170
static constexpr int KP1  = 1184;             // ceil(1170/32)*32
static constexpr int K64  = 64 * 9;           // 576 (multiple of 32)
static constexpr int NCH  = 432;              // conv4 true channels (3*9*16)
static constexpr int NCHP = 448;              // padded to 7 * (4 tiles * 16)
static constexpr int NOFF = 144;              // 16 groups * 9 taps
static constexpr int WAVES = 8;               // waves per 256-thread block
static constexpr int MTILES = B_ * H_ * (W_ / 16);   // 4096

enum { EPI_LRELU = 0, EPI_OFFSET = 1 };

__device__ __forceinline__ float fast_tanh(float x) {
  const float e = __expf(-2.f * fabsf(x));                  // e in (0,1]
  const float t = (1.f - e) * __builtin_amdgcn_rcpf(1.f + e);
  return copysignf(t, x);
}
__device__ __forceinline__ float fast_sigmoid(float x) {
  return __builtin_amdgcn_rcpf(1.f + __expf(-x));
}

// Async DMA of 16 bytes global -> LDS (tracked by ASYNCcnt, not LOADcnt).
// VDST operand carries the LDS byte address = low 32 bits of the generic ptr.
__device__ __forceinline__ void async_b128_to_lds(const _Float16* g, _Float16* l) {
  const unsigned lds_off = (unsigned)(uintptr_t)l;
  const unsigned long long ga = (unsigned long long)(uintptr_t)g;
  asm volatile("global_load_async_to_lds_b128 %0, %1, off"
               :: "v"(lds_off), "v"(ga) : "memory");
}
__device__ __forceinline__ void wait_async_lds() {
  asm volatile("s_wait_asynccnt 0x0" ::: "memory");
}

// ---------------------------------------------------------------------------
// Pack concat(x_flow_warped, x_current, flow) -> f16 feature tensor [B][130][H][W]
// ---------------------------------------------------------------------------
__global__ void pack_feat_kernel(const float* __restrict__ xw,
                                 const float* __restrict__ xc,
                                 const float* __restrict__ fl,
                                 _Float16* __restrict__ feat, int total) {
  int i = blockIdx.x * blockDim.x + threadIdx.x;
  if (i >= total) return;
  int pix = i % HWp;
  int t   = i / HWp;
  int ch  = t % CIN1;
  int b   = t / CIN1;
  float v;
  if (ch < 64)       v = xw[((size_t)(b * 64 + ch)) * HWp + pix];
  else if (ch < 128) v = xc[((size_t)(b * 64 + (ch - 64))) * HWp + pix];
  else               v = fl[((size_t)(b * 2 + (ch - 128))) * HWp + pix];
  feat[i] = (_Float16)v;
}

// ---------------------------------------------------------------------------
// fp32 weight [Otrue][K] (k = c*9 + ky*3 + kx) -> f16 [Opad][Kpad], zero-padded
// ---------------------------------------------------------------------------
__global__ void pack_w_kernel(const float* __restrict__ w, _Float16* __restrict__ wp,
                              int Otrue, int Opad, int K, int Kpad) {
  int i = blockIdx.x * blockDim.x + threadIdx.x;
  if (i >= Opad * Kpad) return;
  int o = i / Kpad;
  int k = i - o * Kpad;
  wp[i] = (o < Otrue && k < K) ? (_Float16)w[(size_t)o * K + k] : (_Float16)0.f;
}

// ---------------------------------------------------------------------------
// Implicit-GEMM 3x3 conv via v_wmma_f32_16x16x32_f16.
// One wave = one 16-pixel M-tile x 4 16-channel N-tiles, K looped in 32-chunks.
// Per chunk the block DMAs the shared 64x32 weight slab to LDS with
// global_load_async_to_lds_b128 (overlapped with im2col VALU staging), then
// both A and B fragments come from LDS.
// EPI_LRELU: leaky-ReLU + f16 store.
// EPI_OFFSET (conv4): fused 10*tanh + flow -> interleaved (dy,dx) f16 pairs,
//                     sigmoid -> f16 mask.  No raw conv4 tensor materialized.
// ---------------------------------------------------------------------------
template <int NT, int EPI>
__global__ __launch_bounds__(256)
void conv3x3_wmma_kernel(const _Float16* __restrict__ in, int Cin, int Ktot, int Kpad,
                         const _Float16* __restrict__ wp, const float* __restrict__ bias,
                         int Cout,
                         _Float16* __restrict__ out16,     // EPI_LRELU
                         const float* __restrict__ flow,   // EPI_OFFSET
                         _Float16* __restrict__ offpair,   // EPI_OFFSET [B][144][HW][2]
                         _Float16* __restrict__ mask16) {  // EPI_OFFSET [B][144][HW]
  __shared__ __attribute__((aligned(64))) _Float16 lds[WAVES][16][32];
  __shared__ __attribute__((aligned(64))) _Float16 wlds[NT * 16][32];  // shared weight slab

  const int tid  = threadIdx.x;
  const int wave = tid >> 5;
  const int lane = tid & 31;
  const int nlo  = lane & 15;
  const int hih  = lane >> 4;                 // 0 / 1 half of the wave

  const int mt  = blockIdx.x * WAVES + wave;  // 0 .. 4095
  const int b   = mt >> 10;                   // / (H * W/16)
  const int rem = mt & 1023;
  const int h   = rem >> 3;
  const int w0  = (rem & 7) << 4;
  const int n0  = (blockIdx.y * NT) << 4;     // block-uniform

  // this thread's 16-byte piece of the per-chunk weight slab
  const int wrow = tid >> 2;                  // 0..63
  const int wseg = (tid & 3) << 3;            // 0,8,16,24 (f16 elements)
  const _Float16* wsrc = wp + (size_t)(n0 + wrow) * Kpad + wseg;
  _Float16* wdst = &wlds[wrow][wseg];

  v8f acc[NT];
#pragma unroll
  for (int t = 0; t < NT; ++t) {
    const int n = n0 + t * 16 + nlo;
    const float bv = (n < Cout) ? bias[n] : 0.f;
#pragma unroll
    for (int i = 0; i < 8; ++i) acc[t][i] = bv;
  }

  const int nChunks = Kpad >> 5;
  for (int kc = 0; kc < nChunks; ++kc) {
    // ---- async DMA: weight slab for this chunk -> LDS (ASYNCcnt) ----
    async_b128_to_lds(wsrc + (kc << 5), wdst);

    // ---- overlapped: stage im2col column k = kc*32 + lane for 16 pixels ----
    const int k = (kc << 5) + lane;
    if (k < Ktot) {
      const int c  = k / 9;
      const int r9 = k - c * 9;
      const int ky = r9 / 3;
      const int kx = r9 - ky * 3;
      const int y  = h + ky - 1;
      const bool rowok = (y >= 0) && (y < H_);
      const _Float16* src = in + ((size_t)(b * Cin + c) * H_ + (rowok ? y : 0)) * W_;
#pragma unroll
      for (int m = 0; m < 16; ++m) {
        const int xx = w0 + m + kx - 1;
        _Float16 v = (_Float16)0.f;
        if (rowok && xx >= 0 && xx < W_) v = src[xx];
        lds[wave][m][lane] = v;
      }
    } else {
#pragma unroll
      for (int m = 0; m < 16; ++m) lds[wave][m][lane] = (_Float16)0.f;
    }
    wait_async_lds();
    __syncthreads();

    // ---- A fragment: M = lane&15; K = 16*half + s (contiguous, matches B) ----
    const _Float16* row = &lds[wave][nlo][0];
    const int k0 = hih ? 16 : 0;
    v8h alo = *(const v8h*)(row + k0);
    v8h ahi = *(const v8h*)(row + k0 + 8);
    v16h a;
#pragma unroll
    for (int i = 0; i < 8; ++i) { a[i] = alo[i]; a[8 + i] = ahi[i]; }

    // ---- B fragments from the LDS weight slab ----
#pragma unroll
    for (int t = 0; t < NT; ++t) {
      v16h bf = *(const v16h*)&wlds[t * 16 + nlo][hih << 4];
      acc[t] = __builtin_amdgcn_wmma_f32_16x16x32_f16(
          false, a, false, bf, (short)0, acc[t], false, false);
    }
    __syncthreads();
  }

  // ---- epilogue ----
  if constexpr (EPI == EPI_LRELU) {
#pragma unroll
    for (int t = 0; t < NT; ++t) {
      const int n = n0 + t * 16 + nlo;
      const size_t base = ((size_t)(b * Cout + n) * H_ + h) * W_ + w0;
#pragma unroll
      for (int r = 0; r < 8; ++r) {
        float v = acc[t][r];
        v = (v >= 0.f) ? v : 0.1f * v;
        out16[base + r + (hih << 3)] = (_Float16)v;
      }
    }
  } else {
    const float* fl0 = flow + ((size_t)(b * 2) * H_ + h) * W_;   // flow ch0
    const float* fl1 = fl0 + (size_t)HWp;                        // flow ch1
#pragma unroll
    for (int t = 0; t < NT; ++t) {
      const int n = n0 + t * 16 + nlo;
      if (n < 288) {
        // offset channel: even -> dy (+= flow1), odd -> dx (+= flow0)
        const int j    = n >> 1;
        const int isdx = n & 1;
        _Float16* dst = offpair + (((size_t)(b * NOFF + j) * H_ + h) * W_) * 2 + isdx;
#pragma unroll
        for (int r = 0; r < 8; ++r) {
          const int w = w0 + r + (hih << 3);
          const float o = 10.f * fast_tanh(acc[t][r]) + (isdx ? fl0[w] : fl1[w]);
          dst[2 * w] = (_Float16)o;
        }
      } else if (n < NCH) {
        _Float16* dst = mask16 + ((size_t)(b * NOFF + (n - 288)) * H_ + h) * W_;
#pragma unroll
        for (int r = 0; r < 8; ++r) {
          dst[w0 + r + (hih << 3)] = (_Float16)fast_sigmoid(acc[t][r]);
        }
      }  // n >= 432: padded channel, dropped
    }
  }
}

// ---------------------------------------------------------------------------
// Deformable conv: bilinear sample with precomputed (dy,dx) pairs and mask,
// staged to LDS and reduced over K = C*9 = 576 with WMMA.  Weights staged per
// chunk via async DMA exactly like the conv kernels.
// ---------------------------------------------------------------------------
__global__ __launch_bounds__(256)
void deform_wmma_kernel(const float* __restrict__ x,
                        const _Float16* __restrict__ offpair,
                        const _Float16* __restrict__ mask16,
                        const _Float16* __restrict__ wpd,
                        const float* __restrict__ bias, float* __restrict__ out) {
  __shared__ __attribute__((aligned(64))) _Float16 lds[WAVES][16][32];
  __shared__ __attribute__((aligned(64))) _Float16 wlds[64][32];

  const int tid  = threadIdx.x;
  const int wave = tid >> 5;
  const int lane = tid & 31;
  const int nlo  = lane & 15;
  const int hih  = lane >> 4;

  const int mt  = blockIdx.x * WAVES + wave;
  const int b   = mt >> 10;
  const int rem = mt & 1023;
  const int h   = rem >> 3;
  const int w0  = (rem & 7) << 4;
  constexpr int NT = 4;                       // 64 output channels

  const int wrow = tid >> 2;
  const int wseg = (tid & 3) << 3;
  const _Float16* wsrc = wpd + (size_t)wrow * K64 + wseg;
  _Float16* wdst = &wlds[wrow][wseg];

  v8f acc[NT];
#pragma unroll
  for (int t = 0; t < NT; ++t) {
    const float bv = bias[t * 16 + nlo];
#pragma unroll
    for (int i = 0; i < 8; ++i) acc[t][i] = bv;
  }

  for (int kc = 0; kc < (K64 >> 5); ++kc) {   // 18 chunks
    async_b128_to_lds(wsrc + (kc << 5), wdst);

    const int kk = (kc << 5) + lane;          // < 576
    const int c  = kk / 9;                    // input channel 0..63
    const int k9 = kk - c * 9;                // 3x3 tap
    const int g  = c >> 2;                    // deform group (Cg = 4)
    const int ky = k9 / 3;
    const int kx = k9 - ky * 3;
    const int j  = g * 9 + k9;

    const _Float16* p_off = offpair + (((size_t)(b * NOFF + j) * H_ + h) * W_) * 2;
    const _Float16* p_mk  = mask16 + ((size_t)(b * NOFF + j) * H_ + h) * W_;
    const float* xc = x + (size_t)(b * C_ + c) * HWp;

#pragma unroll
    for (int m = 0; m < 16; ++m) {
      const int w = w0 + m;
      const float dy = (float)p_off[2 * w];
      const float dx = (float)p_off[2 * w + 1];
      const float py = dy + (float)(h - 1 + ky);               // PAD = 1
      const float px = dx + (float)(w - 1 + kx);
      const float y0f = floorf(py), x0f = floorf(px);
      const int y0 = (int)y0f, x0 = (int)x0f;
      const float wy1 = py - y0f, wx1 = px - x0f;
      auto tap = [&](int yy, int xx) -> float {
        if (yy < 0 || yy >= H_ || xx < 0 || xx >= W_) return 0.f;
        return xc[yy * W_ + xx];
      };
      const float s = tap(y0, x0) * (1.f - wy1) * (1.f - wx1)
                    + tap(y0, x0 + 1) * (1.f - wy1) * wx1
                    + tap(y0 + 1, x0) * wy1 * (1.f - wx1)
                    + tap(y0 + 1, x0 + 1) * wy1 * wx1;
      lds[wave][m][lane] = (_Float16)(s * (float)p_mk[w]);
    }
    wait_async_lds();
    __syncthreads();

    const _Float16* row = &lds[wave][nlo][0];
    const int k0 = hih ? 16 : 0;
    v8h alo = *(const v8h*)(row + k0);
    v8h ahi = *(const v8h*)(row + k0 + 8);
    v16h a;
#pragma unroll
    for (int i = 0; i < 8; ++i) { a[i] = alo[i]; a[8 + i] = ahi[i]; }

#pragma unroll
    for (int t = 0; t < NT; ++t) {
      v16h bf = *(const v16h*)&wlds[t * 16 + nlo][hih << 4];
      acc[t] = __builtin_amdgcn_wmma_f32_16x16x32_f16(
          false, a, false, bf, (short)0, acc[t], false, false);
    }
    __syncthreads();
  }

#pragma unroll
  for (int t = 0; t < NT; ++t) {
    const int n = t * 16 + nlo;
    const size_t base = ((size_t)(b * C_ + n) * H_ + h) * W_ + w0;
#pragma unroll
    for (int r = 0; r < 8; ++r) {
      out[base + r + (hih << 3)] = acc[t][r];
    }
  }
}

// ---------------------------------------------------------------------------
// Launcher
// ---------------------------------------------------------------------------
extern "C" void kernel_launch(void* const* d_in, const int* in_sizes, int n_in,
                              void* d_out, int out_size, void* d_ws, size_t ws_size,
                              hipStream_t stream) {
  const float* x    = (const float*)d_in[0];
  const float* xw   = (const float*)d_in[1];
  const float* xcur = (const float*)d_in[2];
  const float* flow = (const float*)d_in[3];
  const float* w1   = (const float*)d_in[4];
  const float* b1   = (const float*)d_in[5];
  const float* w2   = (const float*)d_in[6];
  const float* b2   = (const float*)d_in[7];
  const float* w3   = (const float*)d_in[8];
  const float* b3   = (const float*)d_in[9];
  const float* w4   = (const float*)d_in[10];
  const float* b4   = (const float*)d_in[11];
  const float* wd   = (const float*)d_in[12];
  const float* bd   = (const float*)d_in[13];

  char* ws = (char*)d_ws;
  auto carve = [&](size_t bytes) -> char* {
    char* p = ws;
    ws += (bytes + 255) & ~(size_t)255;
    return p;
  };
  _Float16* feat = (_Float16*)carve((size_t)B_ * CIN1 * HWp * 2);
  _Float16* hA   = (_Float16*)carve((size_t)B_ * 64 * HWp * 2);
  _Float16* hB   = (_Float16*)carve((size_t)B_ * 64 * HWp * 2);
  _Float16* offp = (_Float16*)carve((size_t)B_ * NOFF * HWp * 2 * 2);  // (dy,dx) pairs
  _Float16* msk  = (_Float16*)carve((size_t)B_ * NOFF * HWp * 2);
  _Float16* wp1  = (_Float16*)carve((size_t)64 * KP1 * 2);
  _Float16* wp2  = (_Float16*)carve((size_t)64 * K64 * 2);
  _Float16* wp3  = (_Float16*)carve((size_t)64 * K64 * 2);
  _Float16* wp4  = (_Float16*)carve((size_t)NCHP * K64 * 2);
  _Float16* wpd  = (_Float16*)carve((size_t)64 * K64 * 2);

  // pack activations + weights to f16
  {
    const int total = B_ * CIN1 * HWp;
    pack_feat_kernel<<<(total + 255) / 256, 256, 0, stream>>>(xw, xcur, flow, feat, total);
  }
  pack_w_kernel<<<(64 * KP1 + 255) / 256, 256, 0, stream>>>(w1, wp1, 64, 64, K1, KP1);
  pack_w_kernel<<<(64 * K64 + 255) / 256, 256, 0, stream>>>(w2, wp2, 64, 64, K64, K64);
  pack_w_kernel<<<(64 * K64 + 255) / 256, 256, 0, stream>>>(w3, wp3, 64, 64, K64, K64);
  pack_w_kernel<<<(NCHP * K64 + 255) / 256, 256, 0, stream>>>(w4, wp4, NCH, NCHP, K64, K64);
  pack_w_kernel<<<(64 * K64 + 255) / 256, 256, 0, stream>>>(wd, wpd, 64, 64, K64, K64);

  const dim3 blk(256);
  const dim3 g1(MTILES / WAVES, 1);           // 512 blocks, 4 N-tiles per wave
  conv3x3_wmma_kernel<4, EPI_LRELU><<<g1, blk, 0, stream>>>(
      feat, CIN1, K1, KP1, wp1, b1, 64, hA, nullptr, nullptr, nullptr);
  conv3x3_wmma_kernel<4, EPI_LRELU><<<g1, blk, 0, stream>>>(
      hA, 64, K64, K64, wp2, b2, 64, hB, nullptr, nullptr, nullptr);
  conv3x3_wmma_kernel<4, EPI_LRELU><<<g1, blk, 0, stream>>>(
      hB, 64, K64, K64, wp3, b3, 64, hA, nullptr, nullptr, nullptr);
  const dim3 g4(MTILES / WAVES, NCHP / 64);   // 448 ch = 7 * (4 tiles * 16)
  conv3x3_wmma_kernel<4, EPI_OFFSET><<<g4, blk, 0, stream>>>(
      hA, 64, K64, K64, wp4, b4, NCH, nullptr, flow, offp, msk);

  deform_wmma_kernel<<<g1, blk, 0, stream>>>(x, offp, msk, wpd, bd, (float*)d_out);
}
